// AttentionLayer_54631984005169
// MI455X (gfx1250) — matrix operand order repaired and checked by reading, not compiled, not run on hardware
//
#include <hip/hip_runtime.h>

typedef __attribute__((ext_vector_type(16))) _Float16 v16h;
typedef __attribute__((ext_vector_type(8)))  float    v8f;

#define B_   8
#define TQ   2048
#define TK   2048
#define DD   512
#define DOUT 1024
#define MBLK 32      // decoder (query) rows per workgroup
#define NBLK 32      // encoder (key) rows per iteration

__global__ __launch_bounds__(256)
void attn_fused_54631984005169(const float* __restrict__ enc,
                               const float* __restrict__ dec,
                               float* __restrict__ out)
{
  // enc block in f16, transposed (d-major): serves Q·K^T (as K^T) and P·V (as V)
  __shared__ _Float16 sKt[DD][NBLK + 8];
  __shared__ float    sSp[4][2][16][33];   // per-d-group partial scores
  __shared__ _Float16 sP[2][16][40];       // softmaxed probs, f16
  __shared__ float    sM[2][16], sL[2][16], sSc[2][16];

  const int tid  = threadIdx.x;
  const int lane = tid & 31;
  const int wave = tid >> 5;
  const int ln   = lane & 15;     // column / M-row index inside a 16-wide tile
  const int hi   = lane >> 4;     // half-wave select
  const int qsub = wave & 1;      // which 16 queries
  const int dgrp = wave >> 1;     // which 128 output dims
  const int db   = dgrp * 128;

  const int b  = blockIdx.y;
  const int t0 = blockIdx.x * MBLK;

  // ---- init softmax running state ----
  if (tid < 32) { sM[tid >> 4][tid & 15] = -1e30f; sL[tid >> 4][tid & 15] = 0.0f; }

  // ---- pass-through: out[b,t,0:512] = dec[b,t,:] ----
  {
    const float4* din = (const float4*)(dec + ((size_t)b * TQ + t0) * DD);
    #pragma unroll
    for (int i = 0; i < 16; ++i) {
      int idx = tid + 256 * i;            // 4096 float4 = 32 rows x 128
      int r   = idx >> 7;
      int c4  = idx & 127;
      float4 v = din[idx];
      float4* orow = (float4*)(out + ((size_t)b * TQ + t0 + r) * DOUT);
      orow[c4] = v;
    }
  }

  // ---- build Q A-fragments once (A layout for v_wmma_f32_16x16x32_f16) ----
  v16h aq[4];
  {
    const float* qrow = dec + ((size_t)b * TQ + (t0 + qsub * 16 + ln)) * DD;
    #pragma unroll
    for (int kk = 0; kk < 4; ++kk) {
      int c = db + kk * 32;
      #pragma unroll
      for (int v = 0; v < 8; ++v) {
        int k = (v < 4) ? (c + hi * 8 + 2 * v) : (c + 16 + hi * 8 + 2 * (v - 4));
        aq[kk][2 * v]     = (_Float16)qrow[k];
        aq[kk][2 * v + 1] = (_Float16)qrow[k + 1];
      }
    }
  }

  // ---- context accumulators: 8 tiles of 16 queries x 16 dims ----
  v8f oacc[8] = {};

  for (int e0 = 0; e0 < TK; e0 += NBLK) {
    __syncthreads();  // previous iteration's readers of sKt/sP done

    // ---- load encoder block, f32->f16, transpose into LDS ----
    {
      const float4* eb = (const float4*)(enc + ((size_t)b * TK + e0) * DD);
      #pragma unroll
      for (int i = 0; i < 16; ++i) {
        int idx = tid + 256 * i;          // 4096 float4 = 32 rows x 128
        int e   = idx >> 7;
        int d4  = (idx & 127) << 2;
        float4 f = eb[idx];
        sKt[d4 + 0][e] = (_Float16)f.x;
        sKt[d4 + 1][e] = (_Float16)f.y;
        sKt[d4 + 2][e] = (_Float16)f.z;
        sKt[d4 + 3][e] = (_Float16)f.w;
      }
    }
    __syncthreads();

    // ---- S partial = Q · K^T over this wave's 128 dims ----
    v8f sacc[2] = {};
    #pragma unroll
    for (int t = 0; t < 2; ++t) {
      #pragma unroll
      for (int kk = 0; kk < 4; ++kk) {
        int c = db + kk * 32;
        v16h bk;
        #pragma unroll
        for (int v = 0; v < 8; ++v) {
          int d = (v < 4) ? (c + hi * 8 + 2 * v) : (c + 16 + hi * 8 + 2 * (v - 4));
          bk[2 * v]     = sKt[d][t * 16 + ln];
          bk[2 * v + 1] = sKt[d + 1][t * 16 + ln];
        }
        sacc[t] = __builtin_amdgcn_wmma_f32_16x16x32_f16(
            false, aq[kk], false, bk, (short)0, sacc[t], false, false);
      }
    }
    #pragma unroll
    for (int t = 0; t < 2; ++t)
      #pragma unroll
      for (int v = 0; v < 8; ++v)
        sSp[dgrp][qsub][v + 8 * hi][t * 16 + ln] = sacc[t][v];
    __syncthreads();

    // ---- online softmax over this key block (one lane per query row) ----
    if (wave < 2 && lane < 16) {
      const int qs = wave, m = lane;
      float rm = -1e30f;
      for (int j = 0; j < NBLK; ++j) {
        float v = sSp[0][qs][m][j] + sSp[1][qs][m][j]
                + sSp[2][qs][m][j] + sSp[3][qs][m][j];
        rm = fmaxf(rm, v);
      }
      float mo = sM[qs][m];
      float mn = fmaxf(mo, rm);
      float sc = __expf(mo - mn);
      float ls = 0.0f;
      for (int j = 0; j < NBLK; ++j) {
        float v = sSp[0][qs][m][j] + sSp[1][qs][m][j]
                + sSp[2][qs][m][j] + sSp[3][qs][m][j];
        float p = __expf(v - mn);
        ls += p;
        sP[qs][m][j] = (_Float16)p;
      }
      sM[qs][m]  = mn;
      sL[qs][m]  = sL[qs][m] * sc + ls;
      sSc[qs][m] = sc;
    }
    __syncthreads();

    // ---- rescale context accumulators ----
    float scv[8];
    #pragma unroll
    for (int v = 0; v < 8; ++v) scv[v] = sSc[qsub][v + 8 * hi];
    #pragma unroll
    for (int t = 0; t < 8; ++t)
      #pragma unroll
      for (int v = 0; v < 8; ++v) oacc[t][v] *= scv[v];

    // ---- P A-fragment ----
    v16h pa;
    #pragma unroll
    for (int v = 0; v < 8; ++v) {
      int k = (v < 4) ? (hi * 8 + 2 * v) : (16 + hi * 8 + 2 * (v - 4));
      pa[2 * v]     = sP[qsub][ln][k];
      pa[2 * v + 1] = sP[qsub][ln][k + 1];
    }

    // ---- context += P · V ----
    #pragma unroll
    for (int tt = 0; tt < 8; ++tt) {
      int d = db + tt * 16 + ln;
      v16h bv;
      #pragma unroll
      for (int v = 0; v < 8; ++v) {
        int e = (v < 4) ? (hi * 8 + 2 * v) : (16 + hi * 8 + 2 * (v - 4));
        bv[2 * v]     = sKt[d][e];
        bv[2 * v + 1] = sKt[d][e + 1];
      }
      oacc[tt] = __builtin_amdgcn_wmma_f32_16x16x32_f16(
          false, pa, false, bv, (short)0, oacc[tt], false, false);
    }
  }

  // ---- finalize: divide by softmax denominator, store context half ----
  float linv[8];
  #pragma unroll
  for (int v = 0; v < 8; ++v) linv[v] = 1.0f / sL[qsub][v + 8 * hi];
  #pragma unroll
  for (int tt = 0; tt < 8; ++tt) {
    int d = db + tt * 16 + ln;
    #pragma unroll
    for (int v = 0; v < 8; ++v) {
      int t = t0 + qsub * 16 + v + 8 * hi;
      out[((size_t)b * TQ + t) * DOUT + DD + d] = oacc[tt][v] * linv[v];
    }
  }
}

extern "C" void kernel_launch(void* const* d_in, const int* in_sizes, int n_in,
                              void* d_out, int out_size, void* d_ws, size_t ws_size,
                              hipStream_t stream) {
  (void)in_sizes; (void)n_in; (void)out_size; (void)d_ws; (void)ws_size;
  const float* enc = (const float*)d_in[0];   // [8, 2048, 512] f32
  const float* dec = (const float*)d_in[1];   // [8, 2048, 512] f32
  float*       out = (float*)d_out;           // [8, 2048, 1024] f32
  dim3 grid(TQ / MBLK, B_);
  attn_fused_54631984005169<<<grid, 256, 0, stream>>>(enc, dec, out);
}
